// EdgeTransformerEncoder_GATv2_13348758355984
// MI455X (gfx1250) — matrix operand order repaired and checked by reading, not compiled, not run on hardware
//
#include <hip/hip_runtime.h>
#include <cstdint>

typedef __bf16 bf16_t;
typedef __attribute__((ext_vector_type(16))) __bf16 v16bf;
typedef __attribute__((ext_vector_type(8)))  __bf16 v8bf;
typedef __attribute__((ext_vector_type(4)))  __bf16 v4bf;
typedef __attribute__((ext_vector_type(4)))  float  v4f;
typedef __attribute__((ext_vector_type(8)))  float  v8f;

#define CAT16(lo, hi) __builtin_shufflevector((lo), (hi), 0,1,2,3,4,5,6,7,8,9,10,11,12,13,14,15)

// ---------------------------------------------------------------------------
// Utility kernels
// ---------------------------------------------------------------------------
__global__ void cvt_bf16_k(const float* __restrict__ s, bf16_t* __restrict__ d, long n) {
  long i = ((long)blockIdx.x * 256 + threadIdx.x) * 4;
  if (i + 3 < n) {
    v4f v = *(const v4f*)(s + i);
    v4bf o = { (bf16_t)v.x, (bf16_t)v.y, (bf16_t)v.z, (bf16_t)v.w };
    *(v4bf*)(d + i) = o;
  } else {
    for (long j = i; j < n; ++j) d[j] = (bf16_t)s[j];
  }
}

// dst[c][r] = src[r][c]  (R x C f32 -> C x R bf16)
__global__ void transpose_bf16_k(const float* __restrict__ s, bf16_t* __restrict__ d,
                                 int R, int C) {
  long i = (long)blockIdx.x * 256 + threadIdx.x;
  if (i < (long)R * C) {
    int r = (int)(i / C), c = (int)(i % C);
    d[(long)c * R + r] = (bf16_t)s[(long)r * C + c];
  }
}

__global__ void fill_f32_k(float* p, float v, long n) {
  long i = (long)blockIdx.x * 256 + threadIdx.x;
  if (i < n) p[i] = v;
}

__global__ void rows_bias_k(float* out, const float* __restrict__ bias, long n, int D) {
  long i = (long)blockIdx.x * 256 + threadIdx.x;
  if (i < n) out[i] = bias[i % D];
}

// ---------------------------------------------------------------------------
// bf16 WMMA GEMM: C[M,N] = A[M,K] @ Bt[N,K]^T (+bias) (opt relu). K % 32 == 0.
// Bt is the transposed (N x K) weight so every fragment is contiguous.
// Tile 128x64 per 256-thread WG (8 waves, each wave a 16x64 strip).
// Double-buffered LDS: one barrier per k-step; next tile's global loads are
// issued before the WMMAs so VMEM latency overlaps matrix math.
// GUARD=false (tile-aligned M,N) compiles with zero bounds checks.
// ---------------------------------------------------------------------------
template <bool GUARD>
__global__ __launch_bounds__(256) void gemm_bf16_k(
    const bf16_t* __restrict__ A, const bf16_t* __restrict__ Bt,
    const float* __restrict__ bias, float* __restrict__ C,
    int M, int N, int K, int relu)
{
  __shared__ bf16_t As[2][128 * 32];
  __shared__ bf16_t Bs[2][64 * 32];
  const int tid = threadIdx.x;
  const int wid = tid >> 5;
  const int lane = tid & 31;
  const int bm = blockIdx.x * 128;
  const int bn = blockIdx.y * 64;
  const int wr = wid * 16;               // wave row offset in tile
  const int r = lane & 15, kh = lane >> 4;

  const int arow = tid >> 1, aseg = (tid & 1) * 16;  // A loader coords
  const int brow = tid >> 2, bcol = (tid & 3) * 8;   // B loader coords
  const bool aIn = !GUARD || (bm + arow < M);
  const bool bIn = !GUARD || (bn + brow < N);
  const bf16_t* aPtr = A + (long)(bm + arow) * K + aseg;
  const bf16_t* bPtr = Bt + (long)(bn + brow) * K + bcol;

  const v8bf z8 = {};
  v8f acc[4];
  v8f zacc = {};
#pragma unroll
  for (int c = 0; c < 4; ++c) acc[c] = zacc;

  // prologue: stage tile 0 into buffer 0
  {
    v8bf a0 = z8, a1 = z8, bv = z8;
    if (aIn) { a0 = *(const v8bf*)aPtr; a1 = *(const v8bf*)(aPtr + 8); }
    if (bIn) bv = *(const v8bf*)bPtr;
    *(v8bf*)&As[0][arow * 32 + aseg]     = a0;
    *(v8bf*)&As[0][arow * 32 + aseg + 8] = a1;
    *(v8bf*)&Bs[0][brow * 32 + bcol]     = bv;
  }
  __syncthreads();

  const int T = K >> 5;
  for (int kt = 0; kt < T; ++kt) {
    const int buf = kt & 1;
    const bool more = (kt + 1) < T;
    // issue next tile's global loads early (latency overlaps WMMAs below)
    v8bf na0 = z8, na1 = z8, nbv = z8;
    if (more) {
      const bf16_t* ap = aPtr + (kt + 1) * 32;
      const bf16_t* bp = bPtr + (kt + 1) * 32;
      if (aIn) { na0 = *(const v8bf*)ap; na1 = *(const v8bf*)(ap + 8); }
      if (bIn) nbv = *(const v8bf*)bp;
      __builtin_prefetch((const void*)(ap + 64), 0, 1);  // tile kt+3 -> L2
      __builtin_prefetch((const void*)(bp + 64), 0, 1);
    }

    // A frag: lane r = row, K runs {kh*8..+8} and {16+kh*8..+8} (ISA 16-bit A layout)
    v8bf alo = *(const v8bf*)&As[buf][(wr + r) * 32 + kh * 8];
    v8bf ahi = *(const v8bf*)&As[buf][(wr + r) * 32 + 16 + kh * 8];
    v16bf af = CAT16(alo, ahi);
#pragma unroll
    for (int c = 0; c < 4; ++c) {
      // B frag: lane r = col, K runs {kh*8..+8} and {16+kh*8..+8} (same order as A)
      v8bf blo = *(const v8bf*)&Bs[buf][(c * 16 + r) * 32 + kh * 8];
      v8bf bhi = *(const v8bf*)&Bs[buf][(c * 16 + r) * 32 + 16 + kh * 8];
      v16bf bf = CAT16(blo, bhi);
      acc[c] = __builtin_amdgcn_wmma_f32_16x16x32_bf16(false, af, false, bf, (short)0, acc[c], false, false);
    }

    if (more) { // stage next tile into the other buffer (safe: last read at kt-1)
      const int nb = buf ^ 1;
      *(v8bf*)&As[nb][arow * 32 + aseg]     = na0;
      *(v8bf*)&As[nb][arow * 32 + aseg + 8] = na1;
      *(v8bf*)&Bs[nb][brow * 32 + bcol]     = nbv;
    }
    __syncthreads();
  }

#pragma unroll
  for (int c = 0; c < 4; ++c) {
    int gn = bn + c * 16 + r;
    if (GUARD && gn >= N) continue;
    float bb = bias ? bias[gn] : 0.0f;
#pragma unroll
    for (int j = 0; j < 8; ++j) {
      int gm = bm + wr + j + 8 * kh;     // C layout: M = j + 8*laneHalf, N = lane&15
      if (GUARD && gm >= M) continue;
      float v = acc[c][j] + bb;
      if (relu) v = v > 0.0f ? v : 0.0f;
      C[(long)gm * N + gn] = v;
    }
  }
}

// ---------------------------------------------------------------------------
// Flash attention (single head, D=256): O = softmax(Q K^T / 16) V
// One wave per 16 query rows. Keys streamed in 32-wide blocks.
// Q,K bf16 [N,256] row-major; Vt bf16 [256,N] (V transposed); O f32 [N,256].
// All fragments are pairs of 16B contiguous loads.
// ---------------------------------------------------------------------------
__global__ __launch_bounds__(32) void flash_attn_k(
    const bf16_t* __restrict__ Q, const bf16_t* __restrict__ Km,
    const bf16_t* __restrict__ Vt, float* __restrict__ O, int N)
{
  constexpr int D = 256;
  __shared__ bf16_t Pls[16 * 32];
  const int lane = threadIdx.x;
  const int r = lane & 15, kh = lane >> 4;
  const int qb = blockIdx.x * 16;

  v16bf qf[8];
#pragma unroll
  for (int c = 0; c < 8; ++c) {
    const bf16_t* qrow = Q + (long)(qb + r) * D + c * 32;
    qf[c] = CAT16(*(const v8bf*)(qrow + kh * 8), *(const v8bf*)(qrow + 16 + kh * 8));
  }
  v8f zero = {};
  v8f acc[16];
#pragma unroll
  for (int nc = 0; nc < 16; ++nc) acc[nc] = zero;
  float m[8], l[8];
#pragma unroll
  for (int j = 0; j < 8; ++j) { m[j] = -3.0e38f; l[j] = 0.0f; }

  const float sc = 0.0625f; // 1/sqrt(256)
  for (int kb = 0; kb < N; kb += 32) {
    v8f s0 = zero, s1 = zero;
#pragma unroll
    for (int c = 0; c < 8; ++c) {
      const bf16_t* k0p = Km + (long)(kb + r) * D + c * 32 + kh * 8;
      const bf16_t* k1p = Km + (long)(kb + 16 + r) * D + c * 32 + kh * 8;
      v16bf b0 = CAT16(*(const v8bf*)k0p, *(const v8bf*)(k0p + 16));
      v16bf b1 = CAT16(*(const v8bf*)k1p, *(const v8bf*)(k1p + 16));
      s0 = __builtin_amdgcn_wmma_f32_16x16x32_bf16(false, qf[c], false, b0, (short)0, s0, false, false);
      s1 = __builtin_amdgcn_wmma_f32_16x16x32_bf16(false, qf[c], false, b1, (short)0, s1, false, false);
    }
    // online softmax per row (row j+8*kh lives in VGPR j across 16 lanes of a half)
#pragma unroll
    for (int j = 0; j < 8; ++j) {
      float v0 = s0[j] * sc, v1 = s1[j] * sc;
      float rmax = fmaxf(v0, v1);
#pragma unroll
      for (int off = 1; off < 16; off <<= 1)
        rmax = fmaxf(rmax, __shfl_xor(rmax, off, 32));
      float mn = fmaxf(m[j], rmax);
      float corr = __expf(m[j] - mn);
      float p0 = __expf(v0 - mn), p1 = __expf(v1 - mn);
      float rs = p0 + p1;
#pragma unroll
      for (int off = 1; off < 16; off <<= 1)
        rs += __shfl_xor(rs, off, 32);
      l[j] = l[j] * corr + rs;
      m[j] = mn;
#pragma unroll
      for (int nc = 0; nc < 16; ++nc) acc[nc][j] *= corr;
      Pls[(j + 8 * kh) * 32 + r]      = (bf16_t)p0;
      Pls[(j + 8 * kh) * 32 + 16 + r] = (bf16_t)p1;
    }
    __syncthreads();
    // reload P as A-fragment (16x32)
    v16bf pf = CAT16(*(const v8bf*)&Pls[r * 32 + kh * 8],
                     *(const v8bf*)&Pls[r * 32 + 16 + kh * 8]);
#pragma unroll
    for (int nc = 0; nc < 16; ++nc) {
      const bf16_t* vrow = Vt + (long)(nc * 16 + r) * N + kb + kh * 8;
      v16bf vf = CAT16(*(const v8bf*)vrow, *(const v8bf*)(vrow + 16));
      acc[nc] = __builtin_amdgcn_wmma_f32_16x16x32_bf16(false, pf, false, vf, (short)0, acc[nc], false, false);
    }
    __syncthreads();
  }
#pragma unroll
  for (int j = 0; j < 8; ++j) {
    float inv = 1.0f / l[j];
    int row = qb + j + 8 * kh;
#pragma unroll
    for (int nc = 0; nc < 16; ++nc)
      O[(long)row * D + nc * 16 + r] = acc[nc][j] * inv;
  }
}

// ---------------------------------------------------------------------------
// Reductions + LayerNorm kernels (one 256-thread block per row)
// ---------------------------------------------------------------------------
__device__ inline float blockReduceSum(float v) {
  __shared__ float sm[8];
#pragma unroll
  for (int off = 16; off > 0; off >>= 1) v += __shfl_xor(v, off, 32);
  int w = threadIdx.x >> 5, lq = threadIdx.x & 31;
  if (lq == 0) sm[w] = v;
  __syncthreads();
  if (w == 0) {
    float t = (lq < 8) ? sm[lq] : 0.0f;
#pragma unroll
    for (int off = 4; off > 0; off >>= 1) t += __shfl_xor(t, off, 32);
    if (lq == 0) sm[0] = t;
  }
  __syncthreads();
  float res = sm[0];
  __syncthreads();
  return res;
}

// y = LN(x + f) * g + b  (f nullable)
__global__ __launch_bounds__(256) void ln_residual_k(
    const float* __restrict__ x, const float* __restrict__ f,
    const float* __restrict__ g, const float* __restrict__ b,
    float* __restrict__ y, int D)
{
  long row = blockIdx.x;
  const float* xr = x + row * D;
  const float* fr = f ? f + row * D : nullptr;
  float s = 0.0f, s2 = 0.0f;
  for (int c = threadIdx.x; c < D; c += 256) {
    float v = xr[c] + (fr ? fr[c] : 0.0f);
    s += v; s2 += v * v;
  }
  s = blockReduceSum(s);
  s2 = blockReduceSum(s2);
  float mu = s / D;
  float inv = rsqrtf(s2 / D - mu * mu + 1e-5f);
  for (int c = threadIdx.x; c < D; c += 256) {
    float v = xr[c] + (fr ? fr[c] : 0.0f);
    y[row * D + c] = (v - mu) * inv * g[c] + b[c];
  }
}

// beta = sigmoid([h, skip, h-skip] @ Wb + bb); y = LN(beta*skip + (1-beta)*h)
__global__ __launch_bounds__(256) void beta_ln_k(
    const float* __restrict__ h, const float* __restrict__ sk,
    const float* __restrict__ Wb, const float* __restrict__ bb,
    const float* __restrict__ g, const float* __restrict__ b,
    float* __restrict__ y, int D)
{
  long row = blockIdx.x;
  const float* hr = h + row * D;
  const float* sr = sk + row * D;
  float t = 0.0f;
  for (int c = threadIdx.x; c < D; c += 256) {
    float hv = hr[c], sv = sr[c];
    t += Wb[c] * hv + Wb[D + c] * sv + Wb[2 * D + c] * (hv - sv);
  }
  t = blockReduceSum(t);
  float beta = 1.0f / (1.0f + __expf(-(t + bb[0])));
  float s = 0.0f, s2 = 0.0f;
  for (int c = threadIdx.x; c < D; c += 256) {
    float z = beta * sr[c] + (1.0f - beta) * hr[c];
    s += z; s2 += z * z;
  }
  s = blockReduceSum(s);
  s2 = blockReduceSum(s2);
  float mu = s / D;
  float inv = rsqrtf(s2 / D - mu * mu + 1e-5f);
  for (int c = threadIdx.x; c < D; c += 256) {
    float z = beta * sr[c] + (1.0f - beta) * hr[c];
    y[row * D + c] = (z - mu) * inv * g[c] + b[c];
  }
}

// ---------------------------------------------------------------------------
// GATv2 edge kernels
// ---------------------------------------------------------------------------
// logit[e,h] = att[h,:] . leaky(xl[src] + xr[dst] + (ea[e] @ We + ebias))[h,:]
__global__ __launch_bounds__(256) void edge_logits_k(
    const float* __restrict__ xl, const float* __restrict__ xr,
    const float* __restrict__ ea, const float* __restrict__ We,
    const float* __restrict__ ebias, const float* __restrict__ att,
    const int* __restrict__ src, const int* __restrict__ dst,
    float* __restrict__ logits, int E, int H, int Do, int de)
{
  int i = blockIdx.x * 256 + threadIdx.x;
  if (i >= E * H) return;
  int e = i / H, h = i % H;
  int Dh = Do / H;
  int sN = src[e], dN = dst[e];
  float eav[16];
  if (de > 0)
    for (int j = 0; j < de; ++j) eav[j] = ea[(long)e * de + j];
  float acc = 0.0f;
  for (int d0 = 0; d0 < Dh; ++d0) {
    int c = h * Dh + d0;
    float v = xl[(long)sN * Do + c] + xr[(long)dN * Do + c];
    if (de > 0) {
      float w = ebias ? ebias[c] : 0.0f;
      for (int j = 0; j < de; ++j) w += eav[j] * We[(long)j * Do + c];
      v += w;
    }
    v = (v > 0.0f) ? v : 0.2f * v;
    acc += att[c] * v;
  }
  logits[i] = acc;
}

__device__ inline void atomicMaxFloat(float* addr, float val) {
  int* ia = (int*)addr;
  int old = __float_as_int(*addr);
  while (__int_as_float(old) < val) {
    int prev = atomicCAS(ia, old, __float_as_int(val));
    if (prev == old) break;
    old = prev;
  }
}

__global__ void seg_max_k(const float* __restrict__ logits, const int* __restrict__ dst,
                          float* m, int EH, int H) {
  int i = blockIdx.x * 256 + threadIdx.x;
  if (i >= EH) return;
  int e = i / H, h = i % H;
  atomicMaxFloat(&m[dst[e] * H + h], logits[i]);
}

__global__ void seg_exp_k(const float* __restrict__ logits, const float* __restrict__ m,
                          const int* __restrict__ dst, float* __restrict__ ex,
                          float* ssum, int EH, int H) {
  int i = blockIdx.x * 256 + threadIdx.x;
  if (i >= EH) return;
  int e = i / H, h = i % H;
  float mm = m[dst[e] * H + h];
  if (mm < -1.0e37f) mm = 0.0f;
  float v = __expf(logits[i] - mm);
  ex[i] = v;
  atomicAdd(&ssum[dst[e] * H + h], v);
}

__global__ void alpha_div_k(const float* __restrict__ ex, const float* __restrict__ ssum,
                            const int* __restrict__ dst, float* __restrict__ alpha,
                            int EH, int H) {
  int i = blockIdx.x * 256 + threadIdx.x;
  if (i >= EH) return;
  int e = i / H, h = i % H;
  alpha[i] = ex[i] / (ssum[dst[e] * H + h] + 1e-16f);
}

__global__ void scatter_k(const float* __restrict__ xl, const float* __restrict__ alpha,
                          const int* __restrict__ src, const int* __restrict__ dst,
                          float* out, int E, int Do, int H) {
  long i = (long)blockIdx.x * 256 + threadIdx.x;
  if (i >= (long)E * Do) return;
  int e = (int)(i / Do), c = (int)(i % Do);
  int h = c / (Do / H);
  atomicAdd(&out[(long)dst[e] * Do + c],
            xl[(long)src[e] * Do + c] * alpha[(long)e * H + h]);
}

// ---------------------------------------------------------------------------
// Host orchestration
// ---------------------------------------------------------------------------
static inline unsigned cdiv(long a, int b) { return (unsigned)((a + b - 1) / b); }

static void cvt(const float* s, bf16_t* d, long n, hipStream_t st) {
  cvt_bf16_k<<<cdiv(n, 1024), 256, 0, st>>>(s, d, n);
}
// weight convert + transpose: (K x N) f32 -> (N x K) bf16
static void cvtT(const float* s, bf16_t* d, int K, int N, hipStream_t st) {
  transpose_bf16_k<<<cdiv((long)K * N, 256), 256, 0, st>>>(s, d, K, N);
}
static void gemm(const bf16_t* A, const bf16_t* Bt, const float* bias, float* C,
                 int M, int N, int K, int relu, hipStream_t st) {
  dim3 g(cdiv(M, 128), cdiv(N, 64));
  if ((M % 128 == 0) && (N % 64 == 0))
    gemm_bf16_k<false><<<g, 256, 0, st>>>(A, Bt, bias, C, M, N, K, relu);
  else
    gemm_bf16_k<true><<<g, 256, 0, st>>>(A, Bt, bias, C, M, N, K, relu);
}

extern "C" void kernel_launch(void* const* d_in, const int* in_sizes, int n_in,
                              void* d_out, int out_size, void* d_ws, size_t ws_size,
                              hipStream_t stream)
{
  constexpr int N = 4096, E = 131072, INC = 256, H = 4;
  (void)in_sizes; (void)n_in; (void)out_size; (void)ws_size;

  // pytree-flattened inputs: edge_attr, edge_index, params(187 leaves), x
  const float* eattr_in = (const float*)d_in[0];
  const int*   eidx     = (const int*)d_in[1];
  const int*   srcI = eidx;
  const int*   dstI = eidx + E;
  auto P = [&](int i) { return (const float*)d_in[2 + i]; };
  // block leaves: 0 Wbeta,1 We,2 Wl,3 Wr,4 Wskip,5 att,6 b,7 bbeta,8 bskip,9 lng,10 lnb
  auto BLK = [&](int b, int k) { return P(11 * b + k); };
  const float* edge_W = P(33);
  const float* edge_b = P(34);
  // emb leaves: 0 W1,1 W2,2 Wk,3 Wo,4 Wq,5 Wv,6 b1,7 b2,8 bk,9 bo,10 bq,11 bv,
  //             12 ln1b,13 ln1g,14 ln2b,15 ln2g
  auto EMB = [&](int t, int k) { return P(35 + 16 * t + k); };
  const int LOGSTD_BASE = 179, MU_BASE = 183; // head leaves: 0 Wl,1 Wr,2 att,3 b
  const float* xg = P(187);

  // workspace carve
  char* wsp = (char*)d_ws;
  auto carve = [&](size_t bytes) {
    void* p = (void*)wsp;
    wsp += (bytes + 255) & ~(size_t)255;
    return p;
  };
  const size_t NB = (size_t)N * 512;
  float* xcur = (float*)carve(NB * 4);
  float* t1 = (float*)carve(NB * 4);
  float* t2 = (float*)carve(NB * 4);
  float* t3 = (float*)carve(NB * 4);
  float* t4 = (float*)carve(NB * 4);
  float* t5 = (float*)carve(NB * 4);
  bf16_t* actA = (bf16_t*)carve(NB * 2);
  bf16_t* actB = (bf16_t*)carve(NB * 2);
  bf16_t* actC = (bf16_t*)carve(NB * 2);
  bf16_t* wbf  = (bf16_t*)carve((size_t)512 * 512 * 2);
  bf16_t* wbf2 = (bf16_t*)carve((size_t)512 * 512 * 2);
  float* Wc   = (float*)carve((size_t)16 * 512 * 4);
  float* bc   = (float*)carve(512 * 4);
  float* logitsB = (float*)carve((size_t)E * H * 4);
  float* exB     = (float*)carve((size_t)E * H * 4);
  float* alphaA  = (float*)carve((size_t)E * H * 4);
  float* alphaB  = (float*)carve((size_t)E * H * 4);
  float* segm = (float*)carve((size_t)N * H * 4);
  float* segs = (float*)carve((size_t)N * H * 4);

  const int EH = E * H;

  // ---- 9 transformer layers ----
  const float* cur = xg;
  for (int t = 0; t < 9; ++t) {
    long NC = (long)N * INC;
    cvt(cur, actA, NC, stream);
    cvtT(EMB(t, 4), wbf, 256, 256, stream);                   // Wq^T
    gemm(actA, wbf, EMB(t, 10), t1, N, 256, 256, 0, stream);  // q
    cvtT(EMB(t, 2), wbf, 256, 256, stream);                   // Wk^T
    gemm(actA, wbf, EMB(t, 8), t2, N, 256, 256, 0, stream);   // k
    cvtT(EMB(t, 5), wbf, 256, 256, stream);                   // Wv^T
    gemm(actA, wbf, EMB(t, 11), t3, N, 256, 256, 0, stream);  // v
    cvt(t1, actB, NC, stream);                                // q bf16
    cvt(t2, actC, NC, stream);                                // k bf16
    transpose_bf16_k<<<cdiv(NC, 256), 256, 0, stream>>>(t3, actA, N, 256); // Vt bf16
    flash_attn_k<<<N / 16, 32, 0, stream>>>(actB, actC, actA, t4, N);
    cvt(t4, actB, NC, stream);
    cvtT(EMB(t, 3), wbf, 256, 256, stream);                   // Wo^T
    gemm(actB, wbf, EMB(t, 9), t5, N, 256, 256, 0, stream);
    ln_residual_k<<<N, 256, 0, stream>>>(cur, t5, EMB(t, 13), EMB(t, 12), t1, 256);
    cvt(t1, actA, NC, stream);
    cvtT(EMB(t, 0), wbf, 256, 128, stream);                   // W1^T
    gemm(actA, wbf, EMB(t, 6), t2, N, 128, 256, 1, stream);   // relu
    cvt(t2, actB, (long)N * 128, stream);
    cvtT(EMB(t, 1), wbf, 128, 256, stream);                   // W2^T
    gemm(actB, wbf, EMB(t, 7), t3, N, 256, 128, 0, stream);
    ln_residual_k<<<N, 256, 0, stream>>>(t1, t3, EMB(t, 15), EMB(t, 14), xcur, 256);
    cur = xcur;
  }

  // ---- block0 composite edge projection: Wc = edge_W @ We0, bc = edge_b @ We0 ----
  cvt(edge_W, wbf2, 16 * 256, stream);
  cvtT(BLK(0, 1), wbf, 256, 512, stream);                     // We0^T
  gemm(wbf2, wbf, nullptr, Wc, 16, 512, 256, 0, stream);
  cvt(edge_b, wbf2, 256, stream);
  gemm(wbf2, wbf, nullptr, bc, 1, 512, 256, 0, stream);

  // ---- GATv2 former blocks ----
  auto former = [&](const float* xin, int b, int di, int don,
                    const float* ea, int de, const float* Wef, const float* ebias,
                    float* alphaOut) {
    cvt(xin, actA, (long)N * di, stream);
    cvtT(BLK(b, 2), wbf, di, don, stream);
    gemm(actA, wbf, nullptr, t1, N, don, di, 0, stream);      // xl
    cvtT(BLK(b, 3), wbf, di, don, stream);
    gemm(actA, wbf, nullptr, t2, N, don, di, 0, stream);      // xr
    cvtT(BLK(b, 4), wbf, di, don, stream);
    gemm(actA, wbf, BLK(b, 8), t3, N, don, di, 0, stream);    // skip
    edge_logits_k<<<cdiv(EH, 256), 256, 0, stream>>>(
        t1, t2, ea, Wef, ebias, BLK(b, 5), srcI, dstI, logitsB, E, H, don, de);
    fill_f32_k<<<cdiv((long)N * H, 256), 256, 0, stream>>>(segm, -3.0e38f, (long)N * H);
    seg_max_k<<<cdiv(EH, 256), 256, 0, stream>>>(logitsB, dstI, segm, EH, H);
    fill_f32_k<<<cdiv((long)N * H, 256), 256, 0, stream>>>(segs, 0.0f, (long)N * H);
    seg_exp_k<<<cdiv(EH, 256), 256, 0, stream>>>(logitsB, segm, dstI, exB, segs, EH, H);
    alpha_div_k<<<cdiv(EH, 256), 256, 0, stream>>>(exB, segs, dstI, alphaOut, EH, H);
    rows_bias_k<<<cdiv((long)N * don, 256), 256, 0, stream>>>(t4, BLK(b, 6), (long)N * don, don);
    scatter_k<<<cdiv((long)E * don, 256), 256, 0, stream>>>(t1, alphaOut, srcI, dstI, t4, E, don, H);
    beta_ln_k<<<N, 256, 0, stream>>>(t4, t3, BLK(b, 0), BLK(b, 7), BLK(b, 9), BLK(b, 10), xcur, don);
  };

  former(cur, 0, 256, 512, eattr_in, 16, Wc, bc, alphaA); cur = xcur;
  former(cur, 1, 512, 384, alphaA, 4, BLK(1, 1), nullptr, alphaB); cur = xcur;
  former(cur, 2, 384, 256, alphaB, 4, BLK(2, 1), nullptr, alphaA); cur = xcur;

  // ---- mu / logstd GAT heads (no edge features), write directly to d_out ----
  float* outF = (float*)d_out;
  auto headrun = [&](const float* xin, int base, float* outp) {
    cvt(xin, actA, (long)N * 256, stream);
    cvtT(P(base + 0), wbf, 256, 512, stream);
    gemm(actA, wbf, nullptr, t1, N, 512, 256, 0, stream);     // xl
    cvtT(P(base + 1), wbf, 256, 512, stream);
    gemm(actA, wbf, nullptr, t2, N, 512, 256, 0, stream);     // xr
    edge_logits_k<<<cdiv(EH, 256), 256, 0, stream>>>(
        t1, t2, nullptr, nullptr, nullptr, P(base + 2), srcI, dstI, logitsB, E, H, 512, 0);
    fill_f32_k<<<cdiv((long)N * H, 256), 256, 0, stream>>>(segm, -3.0e38f, (long)N * H);
    seg_max_k<<<cdiv(EH, 256), 256, 0, stream>>>(logitsB, dstI, segm, EH, H);
    fill_f32_k<<<cdiv((long)N * H, 256), 256, 0, stream>>>(segs, 0.0f, (long)N * H);
    seg_exp_k<<<cdiv(EH, 256), 256, 0, stream>>>(logitsB, segm, dstI, exB, segs, EH, H);
    alpha_div_k<<<cdiv(EH, 256), 256, 0, stream>>>(exB, segs, dstI, alphaB, EH, H);
    rows_bias_k<<<cdiv((long)N * 512, 256), 256, 0, stream>>>(outp, P(base + 3), (long)N * 512, 512);
    scatter_k<<<cdiv((long)E * 512, 256), 256, 0, stream>>>(t1, alphaB, srcI, dstI, outp, E, 512, H);
  };
  headrun(cur, MU_BASE, outF);
  headrun(cur, LOGSTD_BASE, outF + (long)N * 512);
}
